// EdgeTransitionTri_89464168775795
// MI455X (gfx1250) — compile-verified, matrix operand order checked by reading, not run on hardware
//
#include <hip/hip_runtime.h>

// ---------------------------------------------------------------------------
// Types / helpers
// ---------------------------------------------------------------------------
typedef unsigned short ushort_t;
typedef __attribute__((ext_vector_type(16))) __bf16 v16bf;
typedef __attribute__((ext_vector_type(8)))  float  v8f;

#define DEV static __device__ __forceinline__

DEV ushort_t f2bf(float f) {
  unsigned u = __float_as_uint(f);
  unsigned r = (u + 0x7FFFu + ((u >> 16) & 1u)) >> 16;   // round-to-nearest-even
  return (ushort_t)r;
}
DEV float bf2f(ushort_t h) { return __uint_as_float(((unsigned)h) << 16); }
DEV float sigmoidf_(float x) { return 1.0f / (1.0f + __expf(-x)); }

union FragU { v16bf v; uint4 q[2]; ushort_t h[16]; };

DEV v8f vzero() { v8f z = {0.f,0.f,0.f,0.f,0.f,0.f,0.f,0.f}; return z; }

// Load a 16x32 bf16 fragment (A layout; B uses identical pattern on W^T[n][k])
// tile points at element (row0, k0) of a row-major bf16 matrix with leading dim ld.
DEV v16bf load_frag(const ushort_t* tile, int ld) {
  int lane = threadIdx.x & 31;
  const ushort_t* p = tile + (size_t)(lane & 15) * ld + ((lane >> 4) << 3);
  FragU f;
  f.q[0] = *(const uint4*)(p);        // K offsets +0..7   -> VGPR 0..3
  f.q[1] = *(const uint4*)(p + 16);   // K offsets +16..23 -> VGPR 4..7
  return f.v;
}

DEV v8f wmma_bf16(v16bf a, v16bf b, v8f c) {
  return __builtin_amdgcn_wmma_f32_16x16x32_bf16(false, a, false, b, (short)0, c,
                                                 false, false);
}

// ---------------------------------------------------------------------------
// Generic GEMM: out[M x 128*gridDim.y cols] = epi(A[M x K] @ Wt^T + bias)
// Wt stored [N][K] bf16, staged through LDS in 128-k chunks.
// 8 waves/block, each wave: 16 rows x 128 cols.
// EPI: 0 = bf16 out, 1 = bf16 relu out, 2 = f32 sigmoid out
// ---------------------------------------------------------------------------
template <int EPI>
__global__ void k_gemm8(const ushort_t* __restrict__ A, int K,
                        const ushort_t* __restrict__ Wt,
                        const float* __restrict__ bias,
                        void* __restrict__ out, int ldo) {
  __shared__ ushort_t wlds[128 * 128];   // 32KB weight chunk [n_local][k_local]
  int wave = threadIdx.x >> 5;
  int m0 = (blockIdx.x * 8 + wave) * 16;
  int n0 = blockIdx.y * 128;
  v8f acc[8];
#pragma unroll
  for (int t = 0; t < 8; ++t) acc[t] = vzero();

  for (int kc = 0; kc < K; kc += 128) {
    // cooperative stage of Wt[n0..n0+127][kc..kc+127] (2048 uint4)
    {
      uint4* dst = (uint4*)wlds;
#pragma unroll
      for (int r = 0; r < 8; ++r) {
        int e = threadIdx.x + r * 256;
        int n = e >> 4;               // 16 uint4 per 128-ushort row
        int k2 = (e & 15) * 8;
        dst[e] = *(const uint4*)(Wt + (size_t)(n0 + n) * K + kc + k2);
      }
    }
    __syncthreads();
    if (kc + 128 < K)
      __builtin_prefetch(A + (size_t)m0 * K + kc + 128, 0, 0);
    for (int kk = 0; kk < 128; kk += 32) {
      v16bf a = load_frag(A + (size_t)m0 * K + kc + kk, K);
      v16bf b[8];
#pragma unroll
      for (int t = 0; t < 8; ++t) b[t] = load_frag(wlds + t * 16 * 128 + kk, 128);
#pragma unroll
      for (int t = 0; t < 8; ++t) acc[t] = wmma_bf16(a, b[t], acc[t]);
    }
    __syncthreads();
  }

  int lane = threadIdx.x & 31;
  int cl = lane & 15, ro = (lane >> 4) * 8;
#pragma unroll
  for (int t = 0; t < 8; ++t) {
    int col = n0 + t * 16 + cl;
    float bv = bias ? bias[col] : 0.f;
#pragma unroll
    for (int v = 0; v < 8; ++v) {
      size_t row = (size_t)(m0 + ro + v);
      float val = acc[t][v] + bv;
      if (EPI == 0)      ((ushort_t*)out)[row * ldo + col] = f2bf(val);
      else if (EPI == 1) ((ushort_t*)out)[row * ldo + col] = f2bf(fmaxf(val, 0.f));
      else               ((float*)out)[row * ldo + col]    = sigmoidf_(val);
    }
  }
}

// Residual GEMM: z[map(row)][col] += gate[row][col] * (A@Wt^T + bias), N = 128
__global__ void k_gemm_resid(const ushort_t* __restrict__ A, int K,
                             const ushort_t* __restrict__ Wt,
                             const float* __restrict__ bias,
                             const float* __restrict__ gate,
                             float* __restrict__ z, int transpose_out) {
  __shared__ ushort_t wlds[128 * 128];
  int wave = threadIdx.x >> 5;
  int m0 = (blockIdx.x * 8 + wave) * 16;
  v8f acc[8];
#pragma unroll
  for (int t = 0; t < 8; ++t) acc[t] = vzero();

  for (int kc = 0; kc < K; kc += 128) {
    {
      uint4* dst = (uint4*)wlds;
#pragma unroll
      for (int r = 0; r < 8; ++r) {
        int e = threadIdx.x + r * 256;
        int n = e >> 4;
        int k2 = (e & 15) * 8;
        dst[e] = *(const uint4*)(Wt + (size_t)n * K + kc + k2);
      }
    }
    __syncthreads();
    if (kc + 128 < K)
      __builtin_prefetch(A + (size_t)m0 * K + kc + 128, 0, 0);
    for (int kk = 0; kk < 128; kk += 32) {
      v16bf a = load_frag(A + (size_t)m0 * K + kc + kk, K);
      v16bf b[8];
#pragma unroll
      for (int t = 0; t < 8; ++t) b[t] = load_frag(wlds + t * 16 * 128 + kk, 128);
#pragma unroll
      for (int t = 0; t < 8; ++t) acc[t] = wmma_bf16(a, b[t], acc[t]);
    }
    __syncthreads();
  }

  int lane = threadIdx.x & 31;
  int cl = lane & 15, ro = (lane >> 4) * 8;
#pragma unroll
  for (int t = 0; t < 8; ++t) {
    int col = t * 16 + cl;
    float bv = bias ? bias[col] : 0.f;
#pragma unroll
    for (int v = 0; v < 8; ++v) {
      int row = m0 + ro + v;
      int zr = transpose_out ? ((row & 255) * 256 + (row >> 8)) : row;
      float g = gate ? gate[(size_t)row * 128 + col] : 1.f;
      z[(size_t)zr * 128 + col] += g * (acc[t][v] + bv);
    }
  }
}

// Gated pair projection for triangle-mul:
//   outT[c][idx] = sigmoid(X@Wg + bg) * (X@Wp + bp),  idx = row (or i/k swapped)
// 64 cols per block (blockIdx.y in {0,1}); both weight slices staged in LDS.
__global__ void k_gemm_gatedpair(const ushort_t* __restrict__ A,
                                 const ushort_t* __restrict__ Wgt,
                                 const float* __restrict__ bg,
                                 const ushort_t* __restrict__ Wpt,
                                 const float* __restrict__ bp,
                                 ushort_t* __restrict__ outT, int swap) {
  __shared__ ushort_t wg_lds[64 * 128];   // 16KB
  __shared__ ushort_t wp_lds[64 * 128];   // 16KB
  int wave = threadIdx.x >> 5;
  int m0 = (blockIdx.x * 8 + wave) * 16;
  int n0 = blockIdx.y * 64;
  {
    uint4* dg = (uint4*)wg_lds;
    uint4* dp = (uint4*)wp_lds;
#pragma unroll
    for (int r = 0; r < 4; ++r) {
      int e = threadIdx.x + r * 256;      // 1024 uint4 each
      int n = e >> 4;
      int k2 = (e & 15) * 8;
      dg[e] = *(const uint4*)(Wgt + (size_t)(n0 + n) * 128 + k2);
      dp[e] = *(const uint4*)(Wpt + (size_t)(n0 + n) * 128 + k2);
    }
  }
  __syncthreads();

  v8f ag[4], ap[4];
#pragma unroll
  for (int t = 0; t < 4; ++t) { ag[t] = vzero(); ap[t] = vzero(); }
  for (int kk = 0; kk < 128; kk += 32) {
    v16bf a = load_frag(A + (size_t)m0 * 128 + kk, 128);
    v16bf bgv[4], bpv[4];
#pragma unroll
    for (int t = 0; t < 4; ++t) {
      bgv[t] = load_frag(wg_lds + t * 16 * 128 + kk, 128);
      bpv[t] = load_frag(wp_lds + t * 16 * 128 + kk, 128);
    }
#pragma unroll
    for (int t = 0; t < 4; ++t) {
      ag[t] = wmma_bf16(a, bgv[t], ag[t]);
      ap[t] = wmma_bf16(a, bpv[t], ap[t]);
    }
  }
  int lane = threadIdx.x & 31;
  int cl = lane & 15, ro = (lane >> 4) * 8;
#pragma unroll
  for (int t = 0; t < 4; ++t) {
    int col = n0 + t * 16 + cl;
    float bgb = bg[col], bpb = bp[col];
#pragma unroll
    for (int v = 0; v < 8; ++v) {
      int row = m0 + ro + v;
      int idx = swap ? ((row & 255) * 256 + (row >> 8)) : row;
      float val = sigmoidf_(ag[t][v] + bgb) * (ap[t][v] + bpb);
      outT[(size_t)col * 65536 + idx] = f2bf(val);
    }
  }
}

// Triangle einsum: per channel c, X_c[i][j] = sum_k A_c[i][k] * B_c[j][k]
// Block = (i-tile, c). A-strip (16x256) staged in LDS shared by 8 waves;
// each wave computes two j-tiles (wave, wave+8).
__global__ void k_tri_einsum(const ushort_t* __restrict__ aT,
                             const ushort_t* __restrict__ bT,
                             float* __restrict__ x) {
  __shared__ ushort_t alds[16 * 256];     // 8KB
  int wave = threadIdx.x >> 5;
  int it = blockIdx.x, c = blockIdx.y;
  const ushort_t* Abase = aT + (size_t)c * 65536 + (size_t)it * 16 * 256;
  {
    const uint4* src = (const uint4*)Abase;   // 512 uint4
    uint4* dst = (uint4*)alds;
    dst[threadIdx.x] = src[threadIdx.x];
    dst[threadIdx.x + 256] = src[threadIdx.x + 256];
  }
  __syncthreads();

  const ushort_t* B0 = bT + (size_t)c * 65536 + (size_t)wave * 16 * 256;
  const ushort_t* B1 = B0 + (size_t)8 * 16 * 256;
  v8f acc0 = vzero(), acc1 = vzero();
  for (int kk = 0; kk < 256; kk += 32) {
    v16bf a = load_frag(alds + kk, 256);
    v16bf b0 = load_frag(B0 + kk, 256);
    v16bf b1 = load_frag(B1 + kk, 256);
    acc0 = wmma_bf16(a, b0, acc0);
    acc1 = wmma_bf16(a, b1, acc1);
  }
  int lane = threadIdx.x & 31;
  int cl = lane & 15, ro = (lane >> 4) * 8;
#pragma unroll
  for (int v = 0; v < 8; ++v) {
    int i = it * 16 + ro + v;
    int j0 = wave * 16 + cl;
    int j1 = (wave + 8) * 16 + cl;
    x[(size_t)c * 65536 + (size_t)i * 256 + j0] = acc0[v];
    x[(size_t)c * 65536 + (size_t)i * 256 + j1] = acc1[v];
  }
}

// ---------------------------------------------------------------------------
// LayerNorm kernels
// ---------------------------------------------------------------------------
__global__ void k_ln(const float* __restrict__ zin, const float* __restrict__ w,
                     const float* __restrict__ b, ushort_t* __restrict__ out) {
  int wave = threadIdx.x >> 5, lane = threadIdx.x & 31;
  size_t row = (size_t)blockIdx.x * 8 + wave;
  const float* p = zin + row * 128;
  float v0 = p[lane], v1 = p[lane + 32], v2 = p[lane + 64], v3 = p[lane + 96];
  float s = v0 + v1 + v2 + v3;
  float s2 = v0 * v0 + v1 * v1 + v2 * v2 + v3 * v3;
#pragma unroll
  for (int m = 16; m >= 1; m >>= 1) {
    s += __shfl_xor(s, m, 32);
    s2 += __shfl_xor(s2, m, 32);
  }
  float mu = s * (1.f / 128.f);
  float var = s2 * (1.f / 128.f) - mu * mu;
  float rs = rsqrtf(var + 1e-5f);
  ushort_t* o = out + row * 128;
  o[lane]      = f2bf((v0 - mu) * rs * w[lane]      + b[lane]);
  o[lane + 32] = f2bf((v1 - mu) * rs * w[lane + 32] + b[lane + 32]);
  o[lane + 64] = f2bf((v2 - mu) * rs * w[lane + 64] + b[lane + 64]);
  o[lane + 96] = f2bf((v3 - mu) * rs * w[lane + 96] + b[lane + 96]);
}

__global__ void k_ln_strided(const float* __restrict__ x, const float* __restrict__ w,
                             const float* __restrict__ b, ushort_t* __restrict__ out) {
  size_t m = (size_t)blockIdx.x * blockDim.x + threadIdx.x;
  float s = 0.f, s2 = 0.f;
  for (int c = 0; c < 128; ++c) {
    float v = x[(size_t)c * 65536 + m];
    s += v; s2 += v * v;
  }
  float mu = s * (1.f / 128.f);
  float var = s2 * (1.f / 128.f) - mu * mu;
  float rs = rsqrtf(var + 1e-5f);
  for (int c = 0; c < 128; ++c) {
    float v = x[(size_t)c * 65536 + m];
    out[m * 128 + c] = f2bf((v - mu) * rs * w[c] + b[c]);
  }
}

// ---------------------------------------------------------------------------
// Triangle attention: block = (h, i, j-half). K-strip (256x32) and V-strip
// (32x256) staged in LDS shared across the 8 waves; scores kept in registers;
// softmax via shfl; P reshaped through per-wave LDS staging into A-fragments.
// ---------------------------------------------------------------------------
__global__ void k_attn(const ushort_t* __restrict__ q, const ushort_t* __restrict__ k,
                       const ushort_t* __restrict__ vt, const float* __restrict__ tbp,
                       const float* __restrict__ gatt, ushort_t* __restrict__ obuf) {
  __shared__ ushort_t klds[256 * 32];     // 16KB [kcol][cc]
  __shared__ ushort_t vlds[32 * 256];     // 16KB [cc][kcol]
  __shared__ float stage[8][16 * 32];     // 16KB P staging
  int wave = threadIdx.x >> 5, lane = threadIdx.x & 31;
  int cl = lane & 15, ro = (lane >> 4) * 8;
  int bid = blockIdx.x;
  int jh = bid & 1;
  int hi = bid >> 1;
  int i = hi & 255;
  int h = hi >> 8;
  int j0 = jh * 128 + wave * 16;

  // cooperative stage of K-strip and V-strip
  {
    uint4* dk = (uint4*)klds;            // 1024 uint4, row = 4 uint4
#pragma unroll
    for (int r = 0; r < 4; ++r) {
      int e = threadIdx.x + r * 256;
      int kr = e >> 2;
      int part = (e & 3) * 8;
      dk[e] = *(const uint4*)(k + ((size_t)(i * 256 + kr)) * 128 + h * 32 + part);
    }
    const uint4* sv = (const uint4*)(vt + ((size_t)(h * 256 + i) * 32) * 256);
    uint4* dv = (uint4*)vlds;            // 1024 uint4 linear copy
#pragma unroll
    for (int r = 0; r < 4; ++r) {
      int e = threadIdx.x + r * 256;
      dv[e] = sv[e];
    }
  }
  __syncthreads();

  // ---- scores = Q K^T / sqrt(32) + tb ----
  v16bf qa = load_frag(q + ((size_t)(i * 256 + j0)) * 128 + h * 32, 128);
  v8f s[16];
#pragma unroll
  for (int t = 0; t < 16; ++t) {
    v16bf kb = load_frag(klds + t * 16 * 32, 32);
    s[t] = wmma_bf16(qa, kb, vzero());
  }
  const float scale = 0.17677669529663687f;  // 1/sqrt(32)
#pragma unroll
  for (int t = 0; t < 16; ++t) {
#pragma unroll
    for (int v = 0; v < 8; ++v) {
      int row = j0 + ro + v;
      int col = t * 16 + cl;
      s[t][v] = s[t][v] * scale + tbp[((size_t)row * 256 + col) * 4 + h];
    }
  }

  // ---- softmax over 256 cols (16 tiles x 16 lanes of this half) ----
  float inv[8];
#pragma unroll
  for (int v = 0; v < 8; ++v) {
    float mx = s[0][v];
#pragma unroll
    for (int t = 1; t < 16; ++t) mx = fmaxf(mx, s[t][v]);
#pragma unroll
    for (int m = 8; m >= 1; m >>= 1) mx = fmaxf(mx, __shfl_xor(mx, m, 32));
    float sm = 0.f;
#pragma unroll
    for (int t = 0; t < 16; ++t) {
      s[t][v] = __expf(s[t][v] - mx);
      sm += s[t][v];
    }
#pragma unroll
    for (int m = 8; m >= 1; m >>= 1) sm += __shfl_xor(sm, m, 32);
    inv[v] = 1.f / sm;
  }
#pragma unroll
  for (int t = 0; t < 16; ++t)
#pragma unroll
    for (int v = 0; v < 8; ++v) s[t][v] *= inv[v];

  // ---- O = P V  (chunks of 32 k-cols through LDS staging) ----
  v8f o0 = vzero(), o1 = vzero();
  float* st = stage[wave];
#pragma unroll
  for (int kc = 0; kc < 8; ++kc) {
#pragma unroll
    for (int tt = 0; tt < 2; ++tt) {
      int t = 2 * kc + tt;
#pragma unroll
      for (int v = 0; v < 8; ++v)
        st[(ro + v) * 32 + tt * 16 + cl] = s[t][v];
    }
    FragU pf;
    const float* prow = st + (lane & 15) * 32 + ((lane >> 4) << 3);
#pragma unroll
    for (int e = 0; e < 8; ++e) pf.h[e] = f2bf(prow[e]);
#pragma unroll
    for (int e = 0; e < 8; ++e) pf.h[8 + e] = f2bf(prow[16 + e]);
    v16bf vb0 = load_frag(vlds + kc * 32, 256);
    v16bf vb1 = load_frag(vlds + 16 * 256 + kc * 32, 256);
    o0 = wmma_bf16(pf.v, vb0, o0);
    o1 = wmma_bf16(pf.v, vb1, o1);
  }

  // ---- gate + store ----
#pragma unroll
  for (int v = 0; v < 8; ++v) {
    int j = j0 + ro + v;
    size_t m = (size_t)i * 256 + j;
    float g0 = gatt[m * 128 + h * 32 + cl];
    obuf[m * 128 + h * 32 + cl] = f2bf(o0[v] * g0);
    float g1 = gatt[m * 128 + h * 32 + 16 + cl];
    obuf[m * 128 + h * 32 + 16 + cl] = f2bf(o1[v] * g1);
  }
}

// ---------------------------------------------------------------------------
// Small elementwise / prep kernels
// ---------------------------------------------------------------------------
__global__ void k_prep_w(const float* __restrict__ W, ushort_t* __restrict__ Wt,
                         int K, int N) {
  int idx = blockIdx.x * blockDim.x + threadIdx.x;
  if (idx >= K * N) return;
  int kk = idx % K, n = idx / K;
  Wt[idx] = f2bf(W[(size_t)kk * N + n]);
}

__global__ void k_transpose_z(const float* __restrict__ z, float* __restrict__ zt) {
  size_t idx = (size_t)blockIdx.x * 256 + threadIdx.x;
  int c = idx & 127;
  size_t m = idx >> 7;
  int i = (int)(m >> 8), j = (int)(m & 255);
  zt[((size_t)(j * 256 + i)) * 128 + c] = z[idx];
}

// vt[h][i][cc][kcol] = v[(i*256+kcol)][h*32+cc]
__global__ void k_transpose_v(const ushort_t* __restrict__ v, ushort_t* __restrict__ vt) {
  size_t idx = (size_t)blockIdx.x * 256 + threadIdx.x;  // over 4*256*32*256
  int kcol = idx & 255;
  size_t r = idx >> 8;
  int cc = (int)(r & 31);
  r >>= 5;
  int i = (int)(r & 255);
  int h = (int)(r >> 8);
  vt[idx] = v[((size_t)(i * 256 + kcol)) * 128 + h * 32 + cc];
}

// tbp[m][h] = zl[m] . Wb[:,h]
__global__ void k_tbproj(const ushort_t* __restrict__ zl, const float* __restrict__ Wb,
                         float* __restrict__ tbp) {
  size_t idx = (size_t)blockIdx.x * 256 + threadIdx.x;  // over 65536*4
  int h = idx & 3;
  size_t m = idx >> 2;
  float acc = 0.f;
  for (int c = 0; c < 128; ++c) acc += bf2f(zl[m * 128 + c]) * Wb[c * 4 + h];
  tbp[idx] = acc;
}

// communication stage 1: a = LN(s@W1+b1)/(eps+N*N), b likewise (per row)
__global__ void k_comm_ab(const float* __restrict__ s1, const float* __restrict__ s2,
                          const float* __restrict__ W1, const float* __restrict__ b1,
                          const float* __restrict__ W2, const float* __restrict__ b2,
                          const float* __restrict__ lnw, const float* __restrict__ lnb,
                          float* __restrict__ abuf, float* __restrict__ bbuf) {
  __shared__ float srow[256];
  __shared__ float red[128];
  int m = blockIdx.x, c = threadIdx.x;  // 128 threads
  const float* sp = (m < 128) ? (s1 + (size_t)m * 256) : (s2 + (size_t)(m - 128) * 256);
  srow[c] = sp[c];
  srow[c + 128] = sp[c + 128];
  __syncthreads();
  float d1 = b1[c], d2 = b2[c];
  for (int kk = 0; kk < 256; ++kk) {
    float sv = srow[kk];
    d1 += sv * W1[(size_t)kk * 128 + c];
    d2 += sv * W2[(size_t)kk * 128 + c];
  }
  float mu1, var1, mu2, var2;
  red[c] = d1; __syncthreads();
  for (int off = 64; off >= 1; off >>= 1) { if (c < off) red[c] += red[c + off]; __syncthreads(); }
  mu1 = red[0] * (1.f / 128.f); __syncthreads();
  red[c] = (d1 - mu1) * (d1 - mu1); __syncthreads();
  for (int off = 64; off >= 1; off >>= 1) { if (c < off) red[c] += red[c + off]; __syncthreads(); }
  var1 = red[0] * (1.f / 128.f); __syncthreads();
  red[c] = d2; __syncthreads();
  for (int off = 64; off >= 1; off >>= 1) { if (c < off) red[c] += red[c + off]; __syncthreads(); }
  mu2 = red[0] * (1.f / 128.f); __syncthreads();
  red[c] = (d2 - mu2) * (d2 - mu2); __syncthreads();
  for (int off = 64; off >= 1; off >>= 1) { if (c < off) red[c] += red[c + off]; __syncthreads(); }
  var2 = red[0] * (1.f / 128.f);
  float invn = 1.f / (1e-3f + 65536.f);
  abuf[(size_t)m * 128 + c] = ((d1 - mu1) * rsqrtf(var1 + 1e-5f) * lnw[c] + lnb[c]) * invn;
  bbuf[(size_t)m * 128 + c] = ((d2 - mu2) * rsqrtf(var2 + 1e-5f) * lnw[c] + lnb[c]) * invn;
}

// communication stage 2: rowterm = a@Wo[:128], colterm = b@Wo[128:]
__global__ void k_comm_rowcol(const float* __restrict__ abuf, const float* __restrict__ bbuf,
                              const float* __restrict__ Wo,
                              float* __restrict__ rowterm, float* __restrict__ colterm) {
  __shared__ float ar[128], br[128];
  int m = blockIdx.x, c = threadIdx.x;
  ar[c] = abuf[(size_t)m * 128 + c];
  br[c] = bbuf[(size_t)m * 128 + c];
  __syncthreads();
  float r = 0.f, co = 0.f;
  for (int kk = 0; kk < 128; ++kk) {
    r += ar[kk] * Wo[(size_t)kk * 128 + c];
    co += br[kk] * Wo[(size_t)(128 + kk) * 128 + c];
  }
  rowterm[(size_t)m * 128 + c] = r;
  colterm[(size_t)m * 128 + c] = co;
}

__global__ void k_assemble(const float* __restrict__ z11, const float* __restrict__ z12,
                           const float* __restrict__ z21, const float* __restrict__ z22,
                           const float* __restrict__ rowterm, const float* __restrict__ colterm,
                           const float* __restrict__ bo, float* __restrict__ z) {
  size_t idx = (size_t)blockIdx.x * 256 + threadIdx.x;  // over 65536*128
  int c = idx & 127;
  size_t m = idx >> 7;
  int i = (int)(m >> 8), j = (int)(m & 255);
  int ii = i & 127, jj = j & 127;
  const float* src = (i < 128) ? ((j < 128) ? z11 : z12) : ((j < 128) ? z21 : z22);
  float base = src[((size_t)ii * 128 + jj) * 128 + c];
  z[idx] = base + rowterm[(size_t)i * 128 + c] + colterm[(size_t)j * 128 + c] + bo[c];
}

__global__ void k_split(const float* __restrict__ z, float* __restrict__ out) {
  size_t idx = (size_t)blockIdx.x * 256 + threadIdx.x;  // over 4*128*128*128
  int c = idx & 127;
  size_t r = idx >> 7;
  int jj = (int)(r & 127); r >>= 7;
  int ii = (int)(r & 127); r >>= 7;
  int qd = (int)r;  // 0..3
  int i = (qd >> 1) * 128 + ii;
  int j = (qd & 1) * 128 + jj;
  out[idx] = z[((size_t)(i * 256 + j)) * 128 + c];
}

// ---------------------------------------------------------------------------
// Host launch
// ---------------------------------------------------------------------------
extern "C" void kernel_launch(void* const* d_in, const int* in_sizes, int n_in,
                              void* d_out, int out_size, void* d_ws, size_t ws_size,
                              hipStream_t stream) {
  (void)in_sizes; (void)n_in; (void)out_size; (void)ws_size;
  const float** in = (const float**)d_in;

  // ---- workspace layout (bytes) ----
  char* ws = (char*)d_ws;
  const size_t OFF_Z  = 0;                  // fp32  65536*128
  const size_t OFF_ZL = 33554432;           // bf16  65536*128
  const size_t OFF_A  = 50331648;           // bf16  (aT / q)          [also hidden]
  const size_t OFF_B  = 67108864;           // bf16  (bT / k)
  const size_t OFF_C  = 83886080;           // bf16  (v)
  const size_t OFF_D  = 100663296;          // bf16  (xn / vt)
  const size_t OFF_E  = 117440512;          // bf16  (obuf)
  const size_t OFF_G  = 134217728;          // fp32  gate
  const size_t OFF_X  = 167772160;          // fp32  (einsum x / zt)
  const size_t OFF_TB = 201326592;          // fp32  65536*4
  const size_t OFF_CM = 202375168;          // comm scratch 4x128KB
  const size_t OFF_W  = 202899456;          // bf16 weight pool

  float*    zbuf  = (float*)(ws + OFF_Z);
  ushort_t* zl    = (ushort_t*)(ws + OFF_ZL);
  ushort_t* bufA  = (ushort_t*)(ws + OFF_A);
  ushort_t* bufB  = (ushort_t*)(ws + OFF_B);
  ushort_t* bufC  = (ushort_t*)(ws + OFF_C);
  ushort_t* bufD  = (ushort_t*)(ws + OFF_D);
  ushort_t* bufE  = (ushort_t*)(ws + OFF_E);
  float*    gbuf  = (float*)(ws + OFF_G);
  float*    xbuf  = (float*)(ws + OFF_X);
  float*    tbp   = (float*)(ws + OFF_TB);
  float*    abuf  = (float*)(ws + OFF_CM);
  float*    bbuf  = (float*)(ws + OFF_CM + 131072);
  float*    rowt  = (float*)(ws + OFF_CM + 262144);
  float*    colt  = (float*)(ws + OFF_CM + 393216);
  ushort_t* WP    = (ushort_t*)(ws + OFF_W);
  ushort_t* hbuf  = bufA;  // transition hidden overlaps A..D (65536*512 bf16)

  auto wslot = [&](int s) { return WP + (size_t)s * 16384; };
  ushort_t* W1T = WP + (size_t)22 * 16384;            // [512][128]
  ushort_t* W2T = W1T + 65536;                         // [128][512]

  // ---- input indices (jax pytree: dict keys sorted) ----
  // 0 s1, 1 s2, 2 z11, 3 z12, 4 z21, 5 z22
  // comm 6..13 : W1 W2 Wo b1 b2 bo ln_b ln_w
  // tae 14..23 : Wb Wg Wk Wo Wq Wv bg bo ln_b ln_w
  // tas 24..33 : same order
  // tmi 34..49 : Wag Wap Wbg Wbp Wg Wo bag bap bbg bbp bg bo ln_in_b ln_in_w ln_out_b ln_out_w
  // tmo 50..65 : same order
  // trans 66..71 : W1 W2 b1 b2 ln_b ln_w

  auto prep = [&](const float* W, ushort_t* Wt, int K, int N) {
    k_prep_w<<<(K * N + 255) / 256, 256, 0, stream>>>(W, Wt, K, N);
  };
  // tmo slots 0..5 : ap ag bp bg g o
  prep(in[51], wslot(0), 128, 128); prep(in[50], wslot(1), 128, 128);
  prep(in[53], wslot(2), 128, 128); prep(in[52], wslot(3), 128, 128);
  prep(in[54], wslot(4), 128, 128); prep(in[55], wslot(5), 128, 128);
  // tmi slots 6..11
  prep(in[35], wslot(6), 128, 128); prep(in[34], wslot(7), 128, 128);
  prep(in[37], wslot(8), 128, 128); prep(in[36], wslot(9), 128, 128);
  prep(in[38], wslot(10), 128, 128); prep(in[39], wslot(11), 128, 128);
  // tas slots 12..16 : q k v g o
  prep(in[28], wslot(12), 128, 128); prep(in[26], wslot(13), 128, 128);
  prep(in[29], wslot(14), 128, 128); prep(in[25], wslot(15), 128, 128);
  prep(in[27], wslot(16), 128, 128);
  // tae slots 17..21 : q k v g o
  prep(in[18], wslot(17), 128, 128); prep(in[16], wslot(18), 128, 128);
  prep(in[19], wslot(19), 128, 128); prep(in[15], wslot(20), 128, 128);
  prep(in[17], wslot(21), 128, 128);
  // transition
  prep(in[66], W1T, 128, 512);
  prep(in[67], W2T, 512, 128);

  // ---- communication + assemble z ----
  k_comm_ab<<<256, 128, 0, stream>>>(in[0], in[1], in[6], in[9], in[7], in[10],
                                     in[13], in[12], abuf, bbuf);
  k_comm_rowcol<<<256, 128, 0, stream>>>(abuf, bbuf, in[8], rowt, colt);
  k_assemble<<<32768, 256, 0, stream>>>(in[2], in[3], in[4], in[5], rowt, colt,
                                        in[11], zbuf);

  // ---- triangle multiplication (outgoing then incoming) ----
  auto run_tri_mul = [&](int s0, const float* bag, const float* bap,
                         const float* bbg, const float* bbp,
                         const float* bg, const float* bo,
                         const float* lniw, const float* lnib,
                         const float* lnow, const float* lnob, int incoming) {
    k_ln<<<8192, 256, 0, stream>>>(zbuf, lniw, lnib, zl);
    k_gemm_gatedpair<<<dim3(512, 2), 256, 0, stream>>>(zl, wslot(s0 + 1), bag,
                                                       wslot(s0 + 0), bap, bufA, incoming);
    k_gemm_gatedpair<<<dim3(512, 2), 256, 0, stream>>>(zl, wslot(s0 + 3), bbg,
                                                       wslot(s0 + 2), bbp, bufB, incoming);
    k_gemm8<2><<<dim3(512, 1), 256, 0, stream>>>(zl, 128, wslot(s0 + 4), bg, gbuf, 128);
    k_tri_einsum<<<dim3(16, 128), 256, 0, stream>>>(bufA, bufB, xbuf);
    k_ln_strided<<<256, 256, 0, stream>>>(xbuf, lnow, lnob, bufD);
    k_gemm_resid<<<512, 256, 0, stream>>>(bufD, 128, wslot(s0 + 5), bo, gbuf, zbuf, 0);
  };
  run_tri_mul(0, in[56], in[57], in[58], in[59], in[60], in[61],
              in[63], in[62], in[65], in[64], 0);  // tmo, outgoing
  run_tri_mul(6, in[40], in[41], in[42], in[43], in[44], in[45],
              in[47], in[46], in[49], in[48], 1);  // tmi, incoming

  // ---- triangle attention (starting node, then ending node via transpose) ----
  auto run_att = [&](const float* zsrc, int s0, const float* Wb, const float* bg,
                     const float* bo, const float* lnw, const float* lnb, int tout) {
    k_ln<<<8192, 256, 0, stream>>>(zsrc, lnw, lnb, zl);
    k_gemm8<0><<<dim3(512, 1), 256, 0, stream>>>(zl, 128, wslot(s0 + 0),
                                                 (const float*)nullptr, bufA, 128);  // q
    k_gemm8<0><<<dim3(512, 1), 256, 0, stream>>>(zl, 128, wslot(s0 + 1),
                                                 (const float*)nullptr, bufB, 128);  // k
    k_gemm8<0><<<dim3(512, 1), 256, 0, stream>>>(zl, 128, wslot(s0 + 2),
                                                 (const float*)nullptr, bufC, 128);  // v
    k_gemm8<2><<<dim3(512, 1), 256, 0, stream>>>(zl, 128, wslot(s0 + 3), bg, gbuf, 128);
    k_tbproj<<<1024, 256, 0, stream>>>(zl, Wb, tbp);
    k_transpose_v<<<32768, 256, 0, stream>>>(bufC, bufD);
    k_attn<<<2048, 256, 0, stream>>>(bufA, bufB, bufD, tbp, gbuf, bufE);
    k_gemm_resid<<<512, 256, 0, stream>>>(bufE, 128, wslot(s0 + 4), bo,
                                          (const float*)nullptr, zbuf, tout);
  };
  run_att(zbuf, 12, in[24], in[30], in[31], in[33], in[32], 0);          // tas
  k_transpose_z<<<32768, 256, 0, stream>>>(zbuf, xbuf);                   // zt
  run_att(xbuf, 17, in[14], in[20], in[21], in[23], in[22], 1);          // tae

  // ---- transition MLP ----
  k_ln<<<8192, 256, 0, stream>>>(zbuf, in[71], in[70], zl);
  k_gemm8<1><<<dim3(512, 4), 256, 0, stream>>>(zl, 128, W1T, in[68], hbuf, 512);
  k_gemm_resid<<<512, 256, 0, stream>>>(hbuf, 512, W2T, in[69],
                                        (const float*)nullptr, zbuf, 0);

  // ---- split back into 4 output blocks ----
  k_split<<<32768, 256, 0, stream>>>(zbuf, (float*)d_out);
}
